// Selfattention_75969381532204
// MI455X (gfx1250) — compile-verified, matrix operand order checked by reading, not run on hardware
//
#include <hip/hip_runtime.h>
#include <hip/hip_bf16.h>

// ---------------------------------------------------------------------------
// Self-attention (B=8, C=512, N=2048, D=64) for gfx1250 (MI455X, wave32).
// bf16 WMMA (v_wmma_f32_16x16x32_bf16) with fp32 accumulation, flash-attention
// structure so the 2048x2048 energy matrix never touches HBM.
// Round 3: k_proj keeps 5 accumulators / wave (single K pass, independent
// load->wmma chains so the scheduler can overlap instead of serializing).
// ---------------------------------------------------------------------------

#define BB 8
#define CC 512
#define NN 2048
#define DD 64
#define JTOT 640   // 64 (q) + 64 (k) + 512 (v) fused projection outputs

typedef __attribute__((ext_vector_type(16))) __bf16 v16bf;
typedef __attribute__((ext_vector_type(8)))  __bf16 v8bf;
typedef __attribute__((ext_vector_type(8)))  float  v8f;

static __device__ __forceinline__ __bf16 to_bf16(float f) {
    // round-to-nearest-even f32 -> bf16
    unsigned u = __float_as_uint(f);
    unsigned r = (u + 0x7FFFu + ((u >> 16) & 1u)) >> 16;
    unsigned short s = (unsigned short)r;
    return __builtin_bit_cast(__bf16, s);
}

static __device__ __forceinline__ float nan0(float v) { return (v == v) ? v : 0.0f; }

// Load one 16x32 bf16 WMMA operand fragment (A or B, they share the lane
// pattern per the CDNA5 ISA 16-bit operand tables):
//   lanes 0-15 : row r = lane,    K = {0..7, 16..23}
//   lanes 16-31: row r = lane-16, K = {8..15, 24..31}
// `base` points at (row 0, k 0); rows are `ld` elements apart. Both halves are
// 16-byte aligned -> two b128 loads per lane.
static __device__ __forceinline__ v16bf load_frag16(const __bf16* base, int ld, int lane) {
    const int r = lane & 15;
    const int h = (lane >> 4) & 1;
    const __bf16* p = base + (size_t)r * (size_t)ld;
    v8bf lo = *reinterpret_cast<const v8bf*>(p + h * 8);
    v8bf hi = *reinterpret_cast<const v8bf*>(p + 16 + h * 8);
    v16bf a;
#pragma unroll
    for (int i = 0; i < 8; ++i) { a[i] = lo[i]; a[i + 8] = hi[i]; }
    return a;
}

// ---------------------------------------------------------------------------
// Kernel 1: x (B,C,N) f32 -> Xt (B,N,C) bf16, transposed + nan_to_num.
// 32x32 LDS tile transpose, coalesced on both global sides.
// ---------------------------------------------------------------------------
__global__ void k_transpose_x(const float* __restrict__ x, __bf16* __restrict__ Xt) {
    __shared__ float tile[32][33];
    const int n0 = blockIdx.x * 32;
    const int c0 = blockIdx.y * 32;
    const int b  = blockIdx.z;
    const int tx = threadIdx.x;        // 0..31
    const int ty = threadIdx.y;        // 0..7
#pragma unroll
    for (int i = 0; i < 4; ++i) {
        int c = c0 + ty + i * 8;
        tile[ty + i * 8][tx] = x[((size_t)b * CC + c) * NN + n0 + tx];
    }
    __syncthreads();
#pragma unroll
    for (int i = 0; i < 4; ++i) {
        int r = ty + i * 8;            // local n
        Xt[((size_t)b * NN + (n0 + r)) * CC + c0 + tx] = to_bf16(nan0(tile[tx][r]));
    }
}

// ---------------------------------------------------------------------------
// Kernel 2: pack Wq (x0.125 to fold the 1/sqrt(D) energy scale), Wk, Wv into
// Wall[640][512] bf16; biases into ball[640] f32 (bq also x0.125).
// ---------------------------------------------------------------------------
__global__ void k_pack_w(const float* __restrict__ Wq, const float* __restrict__ bq,
                         const float* __restrict__ Wk, const float* __restrict__ bk,
                         const float* __restrict__ Wv, const float* __restrict__ bv,
                         __bf16* __restrict__ Wall, float* __restrict__ ball) {
    const int idx = blockIdx.x * 256 + threadIdx.x;
    if (idx < JTOT * CC) {
        const int j = idx >> 9;        // /512
        const int c = idx & 511;
        float v;
        if (j < 64)       v = Wq[j * CC + c] * 0.125f;
        else if (j < 128) v = Wk[(j - 64) * CC + c];
        else              v = Wv[(j - 128) * CC + c];
        Wall[idx] = to_bf16(v);
    }
    if (idx < JTOT) {
        float v = (idx < 64) ? bq[idx] * 0.125f
                             : (idx < 128 ? bk[idx - 64] : bv[idx - 128]);
        ball[idx] = v;
    }
}

// ---------------------------------------------------------------------------
// Kernel 3: fused projection GEMM.  One WG per (b, 16-row n tile).
//   Xs (16x512 bf16, 16KB LDS) staged once, shared by 8 waves.
//   Wave w owns j-tiles {w, w+8, w+16, w+24, w+32}: 5 accumulators, ONE pass
//   over K.  Per K-step: 1 LDS A-frag + 5 independent B-frag load pairs + 5
//   WMMAs -> scheduler can clause the loads and overlap with the matrix ops.
// Outputs: Q[b][n][d] (scaled), Kt[b][n][d], V[b][e][n]  (all bf16).
// ---------------------------------------------------------------------------
__global__ void __launch_bounds__(256)
k_proj(const __bf16* __restrict__ Xt, const __bf16* __restrict__ Wall,
       const float* __restrict__ ball,
       __bf16* __restrict__ Qb, __bf16* __restrict__ Ktb, __bf16* __restrict__ Vb) {
    __shared__ __bf16 Xs[16][CC];      // 16 KB
    const int n0 = blockIdx.x * 16;
    const int b  = blockIdx.y;
    const int tid  = threadIdx.x;
    const int wv   = tid >> 5;
    const int lane = tid & 31;
    const int col  = lane & 15;
    const int hi   = lane >> 4;

    // Stage the 16x512 X tile (contiguous in Xt) with 128-bit copies.
    {
        const uint4* src = reinterpret_cast<const uint4*>(Xt + ((size_t)b * NN + n0) * CC);
        uint4* dst = reinterpret_cast<uint4*>(&Xs[0][0]);
#pragma unroll
        for (int i = 0; i < 4; ++i) dst[tid + i * 256] = src[tid + i * 256];
    }
    __syncthreads();

    // Wave w handles j-tiles w, w+8, w+16, w+24, w+32  (j0 = wv*16 + t*128).
    const __bf16* wb[5];
#pragma unroll
    for (int t = 0; t < 5; ++t)
        wb[t] = Wall + (size_t)(wv * 16 + t * 128) * CC;

    v8f acc[5];
#pragma unroll
    for (int t = 0; t < 5; ++t) acc[t] = (v8f){};

#pragma unroll
    for (int c0 = 0; c0 < CC; c0 += 32) {
        v16bf af = load_frag16(&Xs[0][c0], CC, lane);
        v16bf bf[5];
#pragma unroll
        for (int t = 0; t < 5; ++t) bf[t] = load_frag16(wb[t] + c0, CC, lane);
#pragma unroll
        for (int t = 0; t < 5; ++t)
            acc[t] = __builtin_amdgcn_wmma_f32_16x16x32_bf16(
                false, af, false, bf[t], (short)0, acc[t], false, false);
    }

#pragma unroll
    for (int t = 0; t < 5; ++t) {
        const int j0 = wv * 16 + t * 128;
        const int j  = j0 + col;
        const float bias = ball[j];
#pragma unroll
        for (int vr = 0; vr < 8; ++vr) {
            const int n = n0 + vr + 8 * hi;
            const __bf16 v = to_bf16(acc[t][vr] + bias);
            if (t == 0) {
                // j0 = wv*16: waves 0-3 -> Q columns, waves 4-7 -> Kt columns
                if (j0 < 64) Qb [((size_t)b * NN + n) * DD + j]        = v;
                else         Ktb[((size_t)b * NN + n) * DD + (j - 64)] = v;
            } else {
                Vb[((size_t)b * CC + (j - 128)) * NN + n] = v;
            }
        }
    }
}

// ---------------------------------------------------------------------------
// Kernel 4: flash attention + epilogue.  One WG per (b, 16-query tile m0).
// 8 independent waves, wave w owns output channels [w*64, w*64+64).
// Per 32-key chunk: S = Q.Kt (4 wmma), online softmax, P through per-wave
// LDS, O^T += V.P^T (4 wmma).  K fragments are double-buffered across chunk
// iterations; V fragments issue before the softmax so their latency hides
// behind the exp/shuffle chain.
// ---------------------------------------------------------------------------
__global__ void __launch_bounds__(256)
k_attn(const __bf16* __restrict__ Qb, const __bf16* __restrict__ Ktb,
       const __bf16* __restrict__ Vb, const float* __restrict__ x,
       const float* __restrict__ gammap, float* __restrict__ out) {
    __shared__ __bf16 Pld[8][16][32];  // per-wave P tile (1 KB each)
    __shared__ float  stats[8][16];    // per-wave row-stat broadcast

    const int m0   = blockIdx.x * 16;
    const int b    = blockIdx.y;
    const int wv   = threadIdx.x >> 5;
    const int lane = threadIdx.x & 31;
    const int r    = lane & 15;
    const int h    = lane >> 4;
    const int e0   = wv * 64;

    const __bf16* kbB = Ktb + (size_t)b * NN * DD;
    const __bf16* vbB = Vb + (size_t)b * CC * NN;

    // Q fragments (rows m0..m0+15, K = d chunks of 32), loaded once.
    const __bf16* qbase = Qb + ((size_t)b * NN + m0) * DD;
    v16bf qf[2];
    qf[0] = load_frag16(qbase, DD, lane);
    qf[1] = load_frag16(qbase + 32, DD, lane);

    v8f o[4] = {};
    float mrow[8], lrow[8];
#pragma unroll
    for (int vr = 0; vr < 8; ++vr) { mrow[vr] = -3.0e38f; lrow[vr] = 0.0f; }

    // Preload K fragments for chunk 0 (double buffer across iterations).
    v16bf kf[2][2];
#pragma unroll
    for (int jn = 0; jn < 2; ++jn)
#pragma unroll
        for (int kk = 0; kk < 2; ++kk)
            kf[jn][kk] = load_frag16(kbB + (size_t)(jn * 16) * DD + kk * 32, DD, lane);

    for (int n0 = 0; n0 < NN; n0 += 32) {
        // ---- S = Q . Kt : 16 queries x 32 keys, K-dim = 64 ----
        v8f s[2] = {};
#pragma unroll
        for (int jn = 0; jn < 2; ++jn)
#pragma unroll
            for (int kk = 0; kk < 2; ++kk)
                s[jn] = __builtin_amdgcn_wmma_f32_16x16x32_bf16(
                    false, qf[kk], false, kf[jn][kk], (short)0, s[jn], false, false);

        // Issue this chunk's V fragments now: independent of the softmax,
        // so their latency hides behind the exp/shuffle chain below.
        v16bf vf[4];
#pragma unroll
        for (int t = 0; t < 4; ++t)
            vf[t] = load_frag16(vbB + (size_t)(e0 + t * 16) * NN + n0, NN, lane);

        // Issue next chunk's K fragments (consumed next iteration).
        const int nn = n0 + 32;
        if (nn < NN) {
#pragma unroll
            for (int jn = 0; jn < 2; ++jn)
#pragma unroll
                for (int kk = 0; kk < 2; ++kk)
                    kf[jn][kk] = load_frag16(kbB + (size_t)(nn + jn * 16) * DD + kk * 32,
                                             DD, lane);
            // cache hints two chunks ahead (global_prefetch_b8)
            __builtin_prefetch(kbB + (size_t)(nn + 32 + r) * DD, 0, 1);
            __builtin_prefetch(vbB + (size_t)(e0 + r) * NN + nn + 32, 0, 1);
        }

        // ---- online softmax over this 32-key chunk ----
#pragma unroll
        for (int vr = 0; vr < 8; ++vr) {
            float t = fmaxf(s[0][vr], s[1][vr]);
            t = fmaxf(t, __shfl_xor(t, 1, 32));
            t = fmaxf(t, __shfl_xor(t, 2, 32));
            t = fmaxf(t, __shfl_xor(t, 4, 32));
            t = fmaxf(t, __shfl_xor(t, 8, 32));     // row max within 16-lane half
            const float mn = fmaxf(mrow[vr], t);
            const float sc = __expf(mrow[vr] - mn);
            mrow[vr] = mn;
            const float p0 = __expf(s[0][vr] - mn);
            const float p1 = __expf(s[1][vr] - mn);
            float rs = p0 + p1;
            rs += __shfl_xor(rs, 1, 32);
            rs += __shfl_xor(rs, 2, 32);
            rs += __shfl_xor(rs, 4, 32);
            rs += __shfl_xor(rs, 8, 32);
            lrow[vr] = lrow[vr] * sc + rs;
            // scatter P to the per-wave LDS tile (row = vr+8*h, col = r / 16+r)
            Pld[wv][vr + 8 * h][r]      = to_bf16(p0);
            Pld[wv][vr + 8 * h][16 + r] = to_bf16(p1);
            if (lane == 0)  stats[wv][vr]     = sc;
            if (lane == 16) stats[wv][8 + vr] = sc;
        }
        // wave-private LDS store->load turnaround
        asm volatile("s_wait_dscnt 0x0" ::: "memory");
        const float scm = stats[wv][r];             // rescale per query column
#pragma unroll
        for (int t = 0; t < 4; ++t)
#pragma unroll
            for (int vr = 0; vr < 8; ++vr) o[t][vr] *= scm;

        // P^T as B operand: column m = r -> row m of Pld, K = key index
        v16bf pb = load_frag16(&Pld[wv][0][0], 32, lane);

        // ---- O^T += V . P^T : rows = 16 channels, K-dim = 32 keys ----
#pragma unroll
        for (int t = 0; t < 4; ++t)
            o[t] = __builtin_amdgcn_wmma_f32_16x16x32_bf16(
                false, vf[t], false, pb, (short)0, o[t], false, false);
    }

    // ---- epilogue: normalize, gamma*out + nan_to_num(x) ----
#pragma unroll
    for (int vr = 0; vr < 8; ++vr) {
        if (lane == 0)  stats[wv][vr]     = lrow[vr];
        if (lane == 16) stats[wv][8 + vr] = lrow[vr];
    }
    asm volatile("s_wait_dscnt 0x0" ::: "memory");
    const float invl = 1.0f / stats[wv][r];
    const float g = gammap[0];
#pragma unroll
    for (int t = 0; t < 4; ++t) {
#pragma unroll
        for (int vr = 0; vr < 8; ++vr) {
            const int e = e0 + t * 16 + vr + 8 * h;
            const int m = m0 + r;
            const size_t idx = ((size_t)b * CC + e) * NN + m;
            out[idx] = g * (o[t][vr] * invl) + nan0(x[idx]);
        }
    }
}

// ---------------------------------------------------------------------------
extern "C" void kernel_launch(void* const* d_in, const int* in_sizes, int n_in,
                              void* d_out, int out_size, void* d_ws, size_t ws_size,
                              hipStream_t stream) {
    const float* x     = (const float*)d_in[0];
    const float* Wq    = (const float*)d_in[1];
    const float* bq    = (const float*)d_in[2];
    const float* Wk    = (const float*)d_in[3];
    const float* bk    = (const float*)d_in[4];
    const float* Wv    = (const float*)d_in[5];
    const float* bv    = (const float*)d_in[6];
    const float* gamma = (const float*)d_in[7];
    float* out = (float*)d_out;

    // Workspace partition (256B aligned): Xt, Wall, ball, Q, Kt, V
    auto align_up = [](size_t v) { return (v + 255) & ~size_t(255); };
    char* p = (char*)d_ws;
    __bf16* Xt  = (__bf16*)p; p += align_up((size_t)BB * NN * CC * sizeof(__bf16));
    __bf16* Wal = (__bf16*)p; p += align_up((size_t)JTOT * CC * sizeof(__bf16));
    float*  bal = (float*) p; p += align_up((size_t)JTOT * sizeof(float));
    __bf16* Qb  = (__bf16*)p; p += align_up((size_t)BB * NN * DD * sizeof(__bf16));
    __bf16* Ktb = (__bf16*)p; p += align_up((size_t)BB * NN * DD * sizeof(__bf16));
    __bf16* Vb  = (__bf16*)p; p += align_up((size_t)BB * CC * NN * sizeof(__bf16));
    (void)ws_size; (void)in_sizes; (void)n_in; (void)out_size;

    k_transpose_x<<<dim3(NN / 32, CC / 32, BB), dim3(32, 8), 0, stream>>>(x, Xt);
    k_pack_w<<<dim3((JTOT * CC + 255) / 256), dim3(256), 0, stream>>>(
        Wq, bq, Wk, bk, Wv, bv, Wal, bal);
    k_proj<<<dim3(NN / 16, BB), dim3(256), 0, stream>>>(Xt, Wal, bal, Qb, Ktb, Vb);
    k_attn<<<dim3(NN / 16, BB), dim3(256), 0, stream>>>(Qb, Ktb, Vb, x, gamma, out);
}
